// SGWN_58471684768129
// MI455X (gfx1250) — matrix operand order repaired and checked by reading, not compiled, not run on hardware
//
#include <hip/hip_runtime.h>

typedef __attribute__((ext_vector_type(16))) __bf16 v16bf;
typedef __attribute__((ext_vector_type(8)))  __bf16 v8bf;
typedef __attribute__((ext_vector_type(8)))  float  v8f;

#define N_NODES 10000
#define R_SC    3
#define NNZ_E   80000
#define FIN_D   128
#define NHID_D  1024
#define FC1_D   512
#define OUT_D   10
#define G_B     100
#define EDGES   (R_SC * NNZ_E)
#define BN_EPS  1e-5f

// ---------------------------------------------------------------------------
// elementwise helpers
// ---------------------------------------------------------------------------
__global__ __launch_bounds__(256) void fill0_f32(float* __restrict__ p, size_t n) {
  size_t i = (size_t)blockIdx.x * 256 + threadIdx.x;
  if (i < n) p[i] = 0.0f;
}

__global__ __launch_bounds__(256) void cvt_f32_bf16(const float* __restrict__ in,
                                                    __bf16* __restrict__ out, size_t n) {
  size_t i = (size_t)blockIdx.x * 256 + threadIdx.x;
  if (i < n) out[i] = (__bf16)in[i];
}

// W[K][Nc] f32 -> WT[Nc][K] bf16 (one-time per layer; lets GEMM stage B with
// contiguous-K 16B chunks instead of 2-byte transposing LDS stores)
__global__ __launch_bounds__(256) void cvt_transpose_bf16(const float* __restrict__ in,
                                                          __bf16* __restrict__ out,
                                                          int K, int Nc) {
  int i = blockIdx.x * 256 + threadIdx.x;
  if (i >= K * Nc) return;
  int k = i / Nc, n = i % Nc;
  out[(size_t)n * K + k] = (__bf16)in[i];
}

// ---------------------------------------------------------------------------
// bf16 WMMA GEMM:  C[M,Nc] (f32) = A[M,K] (bf16, row-major) @ BT[Nc,K] (bf16)
// Requires: K % 32 == 0, Nc % 64 == 0. M arbitrary (guarded).
// Block = 256 threads (8 waves), tile 128(M) x 64(N), K-step 32.
// Tiles staged memory->LDS with global_load_async_to_lds_b128 (ASYNCcnt).
// ---------------------------------------------------------------------------
#define LDS_STRIDE 40  // 32 + 8 pad bf16 (80 B rows: 16B-aligned, spreads banks)

__global__ __launch_bounds__(256) void gemm_bf16_wmma(
    const __bf16* __restrict__ A, const __bf16* __restrict__ BT,
    float* __restrict__ C, int M, int K, int Nc)
{
  __shared__ __align__(16) __bf16 As[128][LDS_STRIDE];
  __shared__ __align__(16) __bf16 Bs[64][LDS_STRIDE];

  const int tid   = threadIdx.x;
  const int lane  = tid & 31;
  const int wave  = tid >> 5;
  const int waveM = wave & 3;   // 0..3 -> which 32-row strip
  const int waveN = wave >> 2;  // 0..1 -> which 32-col strip
  const int r     = lane & 15;  // row (A) / col (B/C)
  const int hf    = lane >> 4;  // lane half

  const int m0 = blockIdx.y * 128;
  const int n0 = blockIdx.x * 64;

  const unsigned as_base = (unsigned)(uintptr_t)&As[0][0];
  const unsigned bs_base = (unsigned)(uintptr_t)&Bs[0][0];

  // Zero-fill A-tile rows beyond M once; guarded async loads never touch them.
  {
    int tail = M - m0;  // >= 1 by grid construction
    if (tail < 128) {
      for (int idx = tid; idx < (128 - tail) * LDS_STRIDE; idx += 256)
        As[tail + idx / LDS_STRIDE][idx % LDS_STRIDE] = (__bf16)0.0f;
    }
  }

  v8f acc[2][2] = {};

  for (int k0 = 0; k0 < K; k0 += 32) {
    // Stage A tile 128x32: 512 async 16B chunks, 2 per thread.
    #pragma unroll
    for (int c = 0; c < 2; ++c) {
      int chunk = tid + c * 256;
      int row   = chunk >> 2;
      int koff  = (chunk & 3) * 8;
      int gm    = m0 + row;
      if (gm < M) {
        unsigned lds = as_base + row * (LDS_STRIDE * 2) + koff * 2;
        unsigned long long gp =
            (unsigned long long)(uintptr_t)(A + (size_t)gm * K + k0 + koff);
        asm volatile("global_load_async_to_lds_b128 %0, %1, off"
                     :: "v"(lds), "v"(gp) : "memory");
      }
    }
    // Stage B tile 64x32 (BT rows): 256 async 16B chunks, 1 per thread.
    {
      int row  = tid >> 2;          // 0..63
      int koff = (tid & 3) * 8;
      unsigned lds = bs_base + row * (LDS_STRIDE * 2) + koff * 2;
      unsigned long long gp =
          (unsigned long long)(uintptr_t)(BT + (size_t)(n0 + row) * K + k0 + koff);
      asm volatile("global_load_async_to_lds_b128 %0, %1, off"
                   :: "v"(lds), "v"(gp) : "memory");
    }
    asm volatile("s_wait_asynccnt 0" ::: "memory");
    __syncthreads();

    // A fragment (ISA 16-bit A 16x32): lane half 0 -> K {0..7,16..23}, half 1 -> K {8..15,24..31}
    // B fragment (same K order as A):  lane half 0 -> K {0..7,16..23}, half 1 -> K {8..15,24..31}
    #pragma unroll
    for (int mi = 0; mi < 2; ++mi) {
      union { v16bf v; v8bf h[2]; } a;
      const int arow = waveM * 32 + mi * 16 + r;
      a.h[0] = *(const v8bf*)(&As[arow][hf * 8]);
      a.h[1] = *(const v8bf*)(&As[arow][hf * 8 + 16]);
      #pragma unroll
      for (int ni = 0; ni < 2; ++ni) {
        union { v16bf v; v8bf h[2]; } b;
        const int bcol = waveN * 32 + ni * 16 + r;
        b.h[0] = *(const v8bf*)(&Bs[bcol][hf * 8]);
        b.h[1] = *(const v8bf*)(&Bs[bcol][hf * 8 + 16]);
        acc[mi][ni] = __builtin_amdgcn_wmma_f32_16x16x32_bf16(
            false, a.v, false, b.v, (short)0, acc[mi][ni], false, false);
      }
    }
    __syncthreads();
  }

  // C/D layout: VGPR j -> row j (lanes 0-15) / row 8+j (lanes 16-31); col = lane&15
  const int cn = n0 + waveN * 32 + r;
  #pragma unroll
  for (int mi = 0; mi < 2; ++mi)
    #pragma unroll
    for (int j = 0; j < 8; ++j) {
      int m = m0 + waveM * 32 + mi * 16 + hf * 8 + j;
      if (m < M) {
        float* crow = C + (size_t)m * Nc + cn;
        crow[0]  = acc[mi][0][j];
        crow[16] = acc[mi][1][j];
      }
    }
}

// ---------------------------------------------------------------------------
// Sparse wavelet passes (one block per edge, atomics over feature dim).
// Pass 1: U[rows[e] + k*N, :] += vals[e] * XW[cols[e], :]
// Pass 2: H[rows[e], :]       += vals[e] * filt[cols[e]+k*N] * U[cols[e]+k*N, :]
// ---------------------------------------------------------------------------
__global__ __launch_bounds__(256) void scatter_pass1(
    const int* __restrict__ rows, const int* __restrict__ cols,
    const float* __restrict__ vals, const float* __restrict__ XW,
    float* __restrict__ U, int Fout)
{
  int e = blockIdx.x;
  int k = e / NNZ_E;
  int dst = rows[e] + k * N_NODES;
  int src = cols[e];
  float v = vals[e];
  const float* xr = XW + (size_t)src * Fout;
  float* ur = U + (size_t)dst * Fout;
  for (int f = threadIdx.x; f < Fout; f += 256)
    atomicAdd(&ur[f], v * xr[f]);
}

__global__ __launch_bounds__(256) void scatter_pass2(
    const int* __restrict__ rows, const int* __restrict__ cols,
    const float* __restrict__ vals, const float* __restrict__ filt,
    const float* __restrict__ U, float* __restrict__ H, int Fout)
{
  int e = blockIdx.x;
  int k = e / NNZ_E;
  int src = cols[e] + k * N_NODES;
  int dst = rows[e];
  float s = vals[e] * filt[src];
  const float* ur = U + (size_t)src * Fout;
  float* hr = H + (size_t)dst * Fout;
  for (int f = threadIdx.x; f < Fout; f += 256)
    atomicAdd(&hr[f], s * ur[f]);
}

// ---------------------------------------------------------------------------
// BatchNorm (training mode, biased variance). One block per feature column.
// ---------------------------------------------------------------------------
__global__ __launch_bounds__(256) void bn_stats(const float* __restrict__ X, int M, int C,
                                                float* __restrict__ mean, float* __restrict__ var)
{
  int c = blockIdx.x;
  int tid = threadIdx.x;
  float s = 0.f, s2 = 0.f;
  for (int row = tid; row < M; row += 256) {
    float v = X[(size_t)row * C + c];
    s += v; s2 += v * v;
  }
  __shared__ float sh[256], sh2[256];
  sh[tid] = s; sh2[tid] = s2;
  __syncthreads();
  for (int st = 128; st > 0; st >>= 1) {
    if (tid < st) { sh[tid] += sh[tid + st]; sh2[tid] += sh2[tid + st]; }
    __syncthreads();
  }
  if (tid == 0) {
    float m = sh[0] / (float)M;
    mean[c] = m;
    var[c]  = sh2[0] / (float)M - m * m;
  }
}

template <bool RELU, bool BF16OUT>
__global__ __launch_bounds__(256) void bn_apply(
    const float* __restrict__ X, size_t total, int C,
    const float* __restrict__ mean, const float* __restrict__ var,
    const float* __restrict__ g, const float* __restrict__ be,
    float* __restrict__ Yf, __bf16* __restrict__ Yh)
{
  size_t i = (size_t)blockIdx.x * 256 + threadIdx.x;
  if (i >= total) return;
  int c = (int)(i % (size_t)C);
  float sc = g[c] * rsqrtf(var[c] + BN_EPS);
  float y = (X[i] - mean[c]) * sc + be[c];
  if (RELU) y = fmaxf(y, 0.0f);
  if (BF16OUT) Yh[i] = (__bf16)y; else Yf[i] = y;
}

// ---------------------------------------------------------------------------
// Global mean pool
// ---------------------------------------------------------------------------
__global__ __launch_bounds__(256) void count_nodes(const int* __restrict__ batch,
                                                   float* __restrict__ cnt, int n) {
  int i = blockIdx.x * 256 + threadIdx.x;
  if (i < n) atomicAdd(&cnt[batch[i]], 1.0f);
}

__global__ __launch_bounds__(256) void pool_add(const float* __restrict__ H,
                                                const int* __restrict__ batch,
                                                float* __restrict__ pooled, int C) {
  int nid = blockIdx.x;
  int g = batch[nid];
  const float* hr = H + (size_t)nid * C;
  float* pr = pooled + (size_t)g * C;
  for (int f = threadIdx.x; f < C; f += 256)
    atomicAdd(&pr[f], hr[f]);
}

__global__ __launch_bounds__(256) void pool_div(float* __restrict__ pooled,
                                                const float* __restrict__ cnt,
                                                int C, int G) {
  int i = blockIdx.x * 256 + threadIdx.x;
  if (i < G * C) pooled[i] /= fmaxf(cnt[i / C], 1.0f);
}

// ---------------------------------------------------------------------------
// Tiny FC2 (100x512 @ 512x10) — bandwidth-trivial, scalar kernel.
// ---------------------------------------------------------------------------
__global__ __launch_bounds__(256) void fc_small(const float* __restrict__ A,
                                                const float* __restrict__ W,
                                                float* __restrict__ C,
                                                int M, int K, int Nc) {
  int i = blockIdx.x * 256 + threadIdx.x;
  if (i >= M * Nc) return;
  int m = i / Nc, n = i % Nc;
  float s = 0.f;
  for (int k = 0; k < K; ++k) s += A[(size_t)m * K + k] * W[(size_t)k * Nc + n];
  C[i] = s;
}

static inline unsigned nblk(size_t n) { return (unsigned)((n + 255) / 256); }

// ---------------------------------------------------------------------------
extern "C" void kernel_launch(void* const* d_in, const int* in_sizes, int n_in,
                              void* d_out, int out_size, void* d_ws, size_t ws_size,
                              hipStream_t stream)
{
  const float* x     = (const float*)d_in[0];
  const int*   rows  = (const int*)  d_in[1];
  const int*   cols  = (const int*)  d_in[2];
  const float* vals  = (const float*)d_in[3];
  const int*   batch = (const int*)  d_in[4];
  const float* W1    = (const float*)d_in[5];
  const float* filt1 = (const float*)d_in[6];
  // d_in[7] = b1 : cancelled by training-mode BN (mean subtraction)
  const float* g1    = (const float*)d_in[8];
  const float* be1   = (const float*)d_in[9];
  const float* W2    = (const float*)d_in[10];
  const float* filt2 = (const float*)d_in[11];
  // d_in[12] = b2 : cancelled
  const float* g2    = (const float*)d_in[13];
  const float* be2   = (const float*)d_in[14];
  const float* Wf1   = (const float*)d_in[15];
  // d_in[16] = bf1 : cancelled
  const float* g3    = (const float*)d_in[17];
  const float* be3   = (const float*)d_in[18];
  const float* Wf2   = (const float*)d_in[19];
  // d_in[20] = bf2 : cancelled
  const float* g4    = (const float*)d_in[21];
  const float* be4   = (const float*)d_in[22];

  // ---- workspace carve ----
  char* w = (char*)d_ws;
  size_t off = 0;
  auto carve = [&](size_t bytes) -> void* {
    void* p = (void*)(w + off);
    off += (bytes + 255) & ~(size_t)255;
    return p;
  };
  float*  XW   = (float*) carve((size_t)N_NODES * NHID_D * 4);           // GEMM out / conv in
  float*  U    = (float*) carve((size_t)R_SC * N_NODES * NHID_D * 4);    // wavelet intermediate
  float*  H    = (float*) carve((size_t)N_NODES * NHID_D * 4);           // conv out
  float*  PO   = (float*) carve((size_t)G_B * NHID_D * 4);
  float*  CNT  = (float*) carve((size_t)G_B * 4);
  float*  MEAN = (float*) carve((size_t)NHID_D * 4);
  float*  VAR  = (float*) carve((size_t)NHID_D * 4);
  float*  F1   = (float*) carve((size_t)G_B * FC1_D * 4);
  float*  F2   = (float*) carve((size_t)G_B * OUT_D * 4);
  __bf16* AB   = (__bf16*)carve((size_t)N_NODES * NHID_D * 2);           // bf16 A operand
  __bf16* WT   = (__bf16*)carve((size_t)NHID_D * NHID_D * 2);           // bf16 W^T operand

  const size_t nNH = (size_t)N_NODES * NHID_D;
  const size_t nU  = (size_t)R_SC * N_NODES * NHID_D;

  // ================= Layer 1: sgwconv(x, W1, filt1) =================
  cvt_f32_bf16<<<nblk((size_t)N_NODES * FIN_D), 256, 0, stream>>>(x, AB, (size_t)N_NODES * FIN_D);
  cvt_transpose_bf16<<<nblk((size_t)FIN_D * NHID_D), 256, 0, stream>>>(W1, WT, FIN_D, NHID_D);
  {
    dim3 g(NHID_D / 64, (N_NODES + 127) / 128);
    gemm_bf16_wmma<<<g, 256, 0, stream>>>(AB, WT, XW, N_NODES, FIN_D, NHID_D);
  }
  fill0_f32<<<nblk(nU), 256, 0, stream>>>(U, nU);
  scatter_pass1<<<EDGES, 256, 0, stream>>>(rows, cols, vals, XW, U, NHID_D);
  fill0_f32<<<nblk(nNH), 256, 0, stream>>>(H, nNH);
  scatter_pass2<<<EDGES, 256, 0, stream>>>(rows, cols, vals, filt1, U, H, NHID_D);
  bn_stats<<<NHID_D, 256, 0, stream>>>(H, N_NODES, NHID_D, MEAN, VAR);
  // BN1 + ReLU, emit bf16 directly as next GEMM's A operand
  bn_apply<true, true><<<nblk(nNH), 256, 0, stream>>>(H, nNH, NHID_D, MEAN, VAR, g1, be1, nullptr, AB);

  // ================= Layer 2: sgwconv(h, W2, filt2) =================
  cvt_transpose_bf16<<<nblk((size_t)NHID_D * NHID_D), 256, 0, stream>>>(W2, WT, NHID_D, NHID_D);
  {
    dim3 g(NHID_D / 64, (N_NODES + 127) / 128);
    gemm_bf16_wmma<<<g, 256, 0, stream>>>(AB, WT, XW, N_NODES, NHID_D, NHID_D);
  }
  fill0_f32<<<nblk(nU), 256, 0, stream>>>(U, nU);
  scatter_pass1<<<EDGES, 256, 0, stream>>>(rows, cols, vals, XW, U, NHID_D);
  fill0_f32<<<nblk(nNH), 256, 0, stream>>>(H, nNH);
  scatter_pass2<<<EDGES, 256, 0, stream>>>(rows, cols, vals, filt2, U, H, NHID_D);
  bn_stats<<<NHID_D, 256, 0, stream>>>(H, N_NODES, NHID_D, MEAN, VAR);
  bn_apply<true, false><<<nblk(nNH), 256, 0, stream>>>(H, nNH, NHID_D, MEAN, VAR, g2, be2, H, nullptr);

  // ================= global mean pool =================
  fill0_f32<<<nblk((size_t)G_B * NHID_D), 256, 0, stream>>>(PO, (size_t)G_B * NHID_D);
  fill0_f32<<<nblk((size_t)G_B), 256, 0, stream>>>(CNT, (size_t)G_B);
  count_nodes<<<nblk((size_t)N_NODES), 256, 0, stream>>>(batch, CNT, N_NODES);
  pool_add<<<N_NODES, 256, 0, stream>>>(H, batch, PO, NHID_D);
  pool_div<<<nblk((size_t)G_B * NHID_D), 256, 0, stream>>>(PO, CNT, NHID_D, G_B);

  // ================= FC1 + BN3 + ReLU =================
  cvt_f32_bf16<<<nblk((size_t)G_B * NHID_D), 256, 0, stream>>>(PO, AB, (size_t)G_B * NHID_D);
  cvt_transpose_bf16<<<nblk((size_t)NHID_D * FC1_D), 256, 0, stream>>>(Wf1, WT, NHID_D, FC1_D);
  {
    dim3 g(FC1_D / 64, (G_B + 127) / 128);
    gemm_bf16_wmma<<<g, 256, 0, stream>>>(AB, WT, F1, G_B, NHID_D, FC1_D);
  }
  bn_stats<<<FC1_D, 256, 0, stream>>>(F1, G_B, FC1_D, MEAN, VAR);
  bn_apply<true, false><<<nblk((size_t)G_B * FC1_D), 256, 0, stream>>>(
      F1, (size_t)G_B * FC1_D, FC1_D, MEAN, VAR, g3, be3, F1, nullptr);

  // ================= FC2 + BN4 -> out =================
  fc_small<<<nblk((size_t)G_B * OUT_D), 256, 0, stream>>>(F1, Wf2, F2, G_B, FC1_D, OUT_D);
  bn_stats<<<OUT_D, 256, 0, stream>>>(F2, G_B, OUT_D, MEAN, VAR);
  bn_apply<false, false><<<nblk((size_t)G_B * OUT_D), 256, 0, stream>>>(
      F2, (size_t)G_B * OUT_D, OUT_D, MEAN, VAR, g4, be4, (float*)d_out, nullptr);
}
